// LSTMEncoder_5617817223770
// MI455X (gfx1250) — matrix operand order repaired and checked by reading, not compiled, not run on hardware
//
#include <hip/hip_runtime.h>

// ---------------------------------------------------------------------------
// Fused LSTM encoder for MI455X (gfx1250, wave32, WMMA).
// B=256,T=1024,F=64,H=128. One workgroup owns 16 batch rows and runs the whole
// T-loop. Weights live in REGISTERS as pre-built WMMA B-fragments (loaded once
// via an LDS transpose stage); LDS carries only double-buffered h / x_t tiles,
// so the steady state needs exactly ONE workgroup barrier per timestep.
// Recurrence is independent per batch row -> no inter-workgroup sync ever.
// ---------------------------------------------------------------------------

typedef __bf16 bf16_t;
typedef __bf16 v16bf __attribute__((ext_vector_type(16)));
typedef __bf16 v8bf  __attribute__((ext_vector_type(8)));
typedef float  v8f   __attribute__((ext_vector_type(8)));

#define B_SZ        256
#define T_SZ        1024
#define F_SZ        64
#define H_SZ        128
#define G_SZ        (4 * H_SZ)     // 512 gate columns
#define ROWS_PER_WG 16
#define THREADS     256            // 8 wave32 waves

// LDS layout (bytes):
//   bias f32[512]           @ 0                  2048
//   WhT  bf16[512][128]     @ 2048               131072   (WhT[n][k] = Wh[k][n])
//   WxT  bf16[512][64]      @ 133120             65536    (WxT[n][k] = Wx[k][n])
//   hbuf bf16[2][16][128]   @ 198656             8192     (double buffered)
//   xbuf bf16[2][16][64]    @ 206848             4096     (double buffered)
#define OFF_BIAS  0
#define OFF_WHT   2048
#define OFF_WXT   133120
#define OFF_HBUF  198656
#define OFF_XBUF  206848
#define LDS_BYTES 210944

#define HBUF_HALFS (ROWS_PER_WG * H_SZ)   // 2048 per buffer
#define XBUF_HALFS (ROWS_PER_WG * F_SZ)   // 1024 per buffer

__device__ __forceinline__ float sigmoid_fast(float x) {
    return 1.0f / (1.0f + __expf(-x));
}
__device__ __forceinline__ float tanh_fast(float x) {
    return 2.0f / (1.0f + __expf(-2.0f * x)) - 1.0f;
}

// A-matrix fragment (16x32 bf16), ISA 7.12.2:
//   lanes 0-15 : row m = lane,    halfs 0..7 = K[klo..klo+7], 8..15 = K[klo+16..klo+23]
//   lanes 16-31: row m = lane-16, klo shifted by +8
__device__ __forceinline__ v16bf load_afrag(const bf16_t* base, int row_stride,
                                            int m, int klo) {
    const bf16_t* p = base + m * row_stride + klo;
    v8bf lo = *(const v8bf*)(p);
    v8bf hi = *(const v8bf*)(p + 16);
    return __builtin_shufflevector(lo, hi, 0, 1, 2, 3, 4, 5, 6, 7,
                                          8, 9, 10, 11, 12, 13, 14, 15);
}

__global__ void __launch_bounds__(THREADS, 1)
lstm_fused_kernel(const float* __restrict__ x,
                  const float* __restrict__ Wx,
                  const float* __restrict__ Wh,
                  const float* __restrict__ b,
                  float* __restrict__ out) {
    extern __shared__ char smem[] __attribute__((aligned(32)));
    float*  bias = (float*)(smem + OFF_BIAS);
    bf16_t* WhT  = (bf16_t*)(smem + OFF_WHT);
    bf16_t* WxT  = (bf16_t*)(smem + OFF_WXT);
    bf16_t* hbuf = (bf16_t*)(smem + OFF_HBUF);
    bf16_t* xbuf = (bf16_t*)(smem + OFF_XBUF);

    const int tid  = threadIdx.x;
    const int lane = tid & 31;
    const int wave = tid >> 5;
    const int n15  = lane & 15;       // column within 16-wide tile
    const int hi16 = lane >> 4;       // lane half-group
    const int r0   = blockIdx.x * ROWS_PER_WG;

    // ---- one-time LDS fill: bias, transposed weights (bf16), h0=0, x_0 ----
    for (int i = tid; i < G_SZ; i += THREADS) bias[i] = b[i];
    for (int i = tid; i < G_SZ * H_SZ; i += THREADS) {
        int n = i >> 7, k = i & 127;                 // WhT[n][k] = Wh[k][n]
        WhT[i] = (bf16_t)Wh[k * G_SZ + n];
    }
    for (int i = tid; i < G_SZ * F_SZ; i += THREADS) {
        int n = i >> 6, k = i & 63;                  // WxT[n][k] = Wx[k][n]
        WxT[i] = (bf16_t)Wx[k * G_SZ + n];
    }
    for (int i = tid; i < 2 * HBUF_HALFS; i += THREADS) hbuf[i] = (bf16_t)0.0f;

    const int xrow = tid >> 4;                        // 16 rows x 16 float4 each
    const int xf0  = (tid & 15) * 4;
    const float* xrow_base = x + (size_t)(r0 + xrow) * (T_SZ * F_SZ) + xf0;
    {
        float4 xv = *(const float4*)(xrow_base);      // t = 0
        union { bf16_t h[4]; uint2 u; } pk;
        pk.h[0] = (bf16_t)xv.x; pk.h[1] = (bf16_t)xv.y;
        pk.h[2] = (bf16_t)xv.z; pk.h[3] = (bf16_t)xv.w;
        *(uint2*)(xbuf + xrow * F_SZ + xf0) = pk.u;
    }
    __syncthreads();

    // ---- per-wave constants: wave w owns h columns [w*16, w*16+16) across all
    //      4 gate blocks, so the gate nonlinearity is wave-local. ----
    const int colb = wave * 16;
    float bv[4];
#pragma unroll
    for (int g = 0; g < 4; ++g) bv[g] = bias[g * H_SZ + colb + n15];

    // ---- hoist ALL loop-invariant B-fragments into registers.
    // B-fragment (ISA 7.12.2): lane n holds column n; halfs 0..7 = K[klo..klo+7],
    // 8..15 = K[klo+16..klo+23], lanes 16-31 klo shifted by +8 (same as A).
    v16bf whB[4][4];   // [gate][k-chunk of 32], K = 128
    v16bf wxB[4][2];   // [gate][k-chunk of 32], K = 64
#pragma unroll
    for (int g = 0; g < 4; ++g) {
        const int zcol = g * H_SZ + colb + n15;       // column in [0,512)
#pragma unroll
        for (int kc = 0; kc < 4; ++kc)
            whB[g][kc] = load_afrag(WhT, H_SZ, zcol, kc * 32 + hi16 * 8);
#pragma unroll
        for (int kc = 0; kc < 2; ++kc)
            wxB[g][kc] = load_afrag(WxT, F_SZ, zcol, kc * 32 + hi16 * 8);
    }

    v8f c_state = {0.f, 0.f, 0.f, 0.f, 0.f, 0.f, 0.f, 0.f};
    v8f hreg    = {0.f, 0.f, 0.f, 0.f, 0.f, 0.f, 0.f, 0.f};

    // ---------------------------- time loop --------------------------------
    // Double-buffered h and x: step t reads buffer `cur`, writes buffer `nxt`.
    // No read/write overlap within a step -> a single barrier per step orders
    // step-t writes against step-(t+1) reads.
    for (int t = 0; t < T_SZ; ++t) {
        const int cur = t & 1, nxt = cur ^ 1;

        // A-fragments for this step (h: K=128 -> 4 chunks, x_t: K=64 -> 2)
        const bf16_t* hb = hbuf + cur * HBUF_HALFS;
        const bf16_t* xb = xbuf + cur * XBUF_HALFS;
        v16bf hA[4], xA[2];
#pragma unroll
        for (int kc = 0; kc < 4; ++kc)
            hA[kc] = load_afrag(hb, H_SZ, n15, kc * 32 + hi16 * 8);
#pragma unroll
        for (int kc = 0; kc < 2; ++kc)
            xA[kc] = load_afrag(xb, F_SZ, n15, kc * 32 + hi16 * 8);

        // prefetch x_{t+1} from HBM while WMMAs run
        float4 xn = make_float4(0.f, 0.f, 0.f, 0.f);
        if (t + 1 < T_SZ) xn = *(const float4*)(xrow_base + (t + 1) * F_SZ);

        // 4 independent accumulation chains x 6 WMMAs, all-register B operands
        v8f acc[4];
#pragma unroll
        for (int g = 0; g < 4; ++g) {
            v8f a;
#pragma unroll
            for (int e = 0; e < 8; ++e) a[e] = bv[g];
#pragma unroll
            for (int kc = 0; kc < 2; ++kc)            // x_t @ Wx
                a = __builtin_amdgcn_wmma_f32_16x16x32_bf16(
                        false, xA[kc], false, wxB[g][kc], (short)0, a, false, false);
#pragma unroll
            for (int kc = 0; kc < 4; ++kc)            // h @ Wh
                a = __builtin_amdgcn_wmma_f32_16x16x32_bf16(
                        false, hA[kc], false, whB[g][kc], (short)0, a, false, false);
            acc[g] = a;
        }

        // stage x_{t+1} into the idle buffer (packed 4xbf16 = one b64 store)
        if (t + 1 < T_SZ) {
            union { bf16_t h[4]; uint2 u; } pk;
            pk.h[0] = (bf16_t)xn.x; pk.h[1] = (bf16_t)xn.y;
            pk.h[2] = (bf16_t)xn.z; pk.h[3] = (bf16_t)xn.w;
            *(uint2*)(xbuf + nxt * XBUF_HALFS + xrow * F_SZ + xf0) = pk.u;
        }

        // gates, entirely in registers (i=acc0, f=acc1, g=acc2, o=acc3)
#pragma unroll
        for (int e = 0; e < 8; ++e) {
            float ig = sigmoid_fast(acc[0][e]);
            float fg = sigmoid_fast(acc[1][e]);
            float gg = tanh_fast(acc[2][e]);
            float og = sigmoid_fast(acc[3][e]);
            float cn = fg * c_state[e] + ig * gg;
            c_state[e] = cn;
            hreg[e] = og * tanh_fast(cn);
        }

        // C-layout -> row-major bf16 h in the idle buffer: VGPR r holds row r
        // (+8 for lanes 16-31), lane holds column colb + n15.
        {
            bf16_t* hn = hbuf + nxt * HBUF_HALFS;
            const int col = colb + n15;
#pragma unroll
            for (int r = 0; r < 8; ++r)
                hn[(r + hi16 * 8) * H_SZ + col] = (bf16_t)hreg[r];
        }
        __syncthreads();   // the ONLY barrier per step
    }

    // ---- emit h_T [256,128] f32 ----
    {
        const int col = colb + n15;
#pragma unroll
        for (int r = 0; r < 8; ++r) {
            const int m = r + hi16 * 8;
            out[(size_t)(r0 + m) * H_SZ + col] = hreg[r];
        }
    }
}

extern "C" void kernel_launch(void* const* d_in, const int* in_sizes, int n_in,
                              void* d_out, int out_size, void* d_ws, size_t ws_size,
                              hipStream_t stream) {
    (void)in_sizes; (void)n_in; (void)out_size; (void)d_ws; (void)ws_size;
    const float* x  = (const float*)d_in[0];
    const float* Wx = (const float*)d_in[1];
    const float* Wh = (const float*)d_in[2];
    const float* b  = (const float*)d_in[3];
    float* out = (float*)d_out;

    // >64KB dynamic LDS: raise the limit (CDNA5 WGP has 320KB).
    (void)hipFuncSetAttribute((const void*)lstm_fused_kernel,
                              hipFuncAttributeMaxDynamicSharedMemorySize,
                              LDS_BYTES);

    dim3 grid(B_SZ / ROWS_PER_WG);   // 16 independent workgroups
    dim3 block(THREADS);             // 8 wave32 waves
    lstm_fused_kernel<<<grid, block, LDS_BYTES, stream>>>(x, Wx, Wh, b, out);
}
